// FasterRCNN_51505247813923
// MI455X (gfx1250) — matrix operand order repaired and checked by reading, not compiled, not run on hardware
//
#include <hip/hip_runtime.h>
#include <hip/hip_bf16.h>
#include <math.h>

#define NPROP 22500
#define POSTN 300
#define NCLSD 21
#define CFEAT 512
#define HFEAT 50
#define WFEAT 50
#define OUTP  7
#define DFEAT (CFEAT*OUTP*OUTP)   /* 25088 */
#define HID   4096
#define MPAD  320                 /* 300 padded to 20 M-tiles of 16 */
#define NEGF  (-1e30f)
#define IMGSZ 800.0f

typedef __attribute__((ext_vector_type(16))) __bf16 v16bf;
typedef __attribute__((ext_vector_type(8)))  float  v8f;
typedef __attribute__((ext_vector_type(4)))  float  f4;

static __device__ __forceinline__ unsigned short f2bf_us(float f){
  union{__bf16 b; unsigned short u;} v; v.b = (__bf16)f; return v.u;
}
static __device__ __forceinline__ __bf16 us_as_bf(unsigned short u){
  union{unsigned short u; __bf16 b;} v; v.u = u; return v.b;
}
static __device__ __forceinline__ float bfu_as_f(unsigned short u){
  union{unsigned u; float f;} v; v.u = ((unsigned)u) << 16; return v.f;
}

// ---------------- Stage 0: mask scores + xywh->xyxy -------------------------
__global__ void prep_kernel(const float* __restrict__ prop,
                            const float* __restrict__ scores,
                            float* __restrict__ sc, float* __restrict__ bxyxy){
  int i = blockIdx.x * blockDim.x + threadIdx.x;
  if (i >= NPROP) return;
  float x = prop[i*4+0], y = prop[i*4+1], w = prop[i*4+2], h = prop[i*4+3];
  bxyxy[i*4+0] = x; bxyxy[i*4+1] = y; bxyxy[i*4+2] = x + w; bxyxy[i*4+3] = y + h;
  sc[i] = (w < 16.0f || h < 16.0f) ? NEGF : scores[i];
}

// ---------------- Stage 1: sequential NMS (argmax scan) ---------------------
__global__ __launch_bounds__(1024)
void nms_kernel(const float* __restrict__ bxyxy, const float* __restrict__ prop,
                float* __restrict__ sc, int* __restrict__ keep,
                float* __restrict__ selxywh, float* __restrict__ selxyxy){
  __shared__ float sval[1024];
  __shared__ int   sidx[1024];
  __shared__ float bb[4];
  int tid = threadIdx.x;
  for (int it = 0; it < POSTN; ++it){
    float bv = -3.0e38f; int bi = 0;
    for (int i = tid; i < NPROP; i += 1024){
      float s = sc[i];
      if (s > bv){ bv = s; bi = i; }
    }
    sval[tid] = bv; sidx[tid] = bi;
    __syncthreads();
    for (int st = 512; st > 0; st >>= 1){
      if (tid < st){
        float ov = sval[tid+st]; int oi = sidx[tid+st];
        if (ov > sval[tid] || (ov == sval[tid] && oi < sidx[tid])){
          sval[tid] = ov; sidx[tid] = oi;
        }
      }
      __syncthreads();
    }
    int best = sidx[0];
    if (tid == 0){
      keep[it] = best;
      float x1 = bxyxy[best*4+0], y1 = bxyxy[best*4+1];
      float x2 = bxyxy[best*4+2], y2 = bxyxy[best*4+3];
      bb[0]=x1; bb[1]=y1; bb[2]=x2; bb[3]=y2;
      selxyxy[it*4+0]=x1; selxyxy[it*4+1]=y1; selxyxy[it*4+2]=x2; selxyxy[it*4+3]=y2;
      selxywh[it*4+0]=prop[best*4+0]; selxywh[it*4+1]=prop[best*4+1];
      selxywh[it*4+2]=prop[best*4+2]; selxywh[it*4+3]=prop[best*4+3];
    }
    __syncthreads();
    float bx1=bb[0], by1=bb[1], bx2=bb[2], by2=bb[3];
    float ba = (bx2-bx1)*(by2-by1);
    for (int i = tid; i < NPROP; i += 1024){
      float x1=bxyxy[i*4+0], y1=bxyxy[i*4+1], x2=bxyxy[i*4+2], y2=bxyxy[i*4+3];
      float ix1=fmaxf(bx1,x1), iy1=fmaxf(by1,y1);
      float ix2=fminf(bx2,x2), iy2=fminf(by2,y2);
      float inter = fmaxf(ix2-ix1,0.f)*fmaxf(iy2-iy1,0.f);
      float ar = (x2-x1)*(y2-y1);
      float iou = inter / (ba + ar - inter + 1e-9f);
      if (iou > 0.7f || i == best) sc[i] = NEGF;
    }
    __syncthreads();
  }
}

// ---------------- Stage 2: RoI-Align -> bf16 activation matrix --------------
__global__ __launch_bounds__(256)
void roi_kernel(const float* __restrict__ feat, const float* __restrict__ selxyxy,
                unsigned short* __restrict__ X){
  int roi = blockIdx.x;
  const float scale = 1.0f/16.0f;
  float x1 = selxyxy[roi*4+0]*scale, y1 = selxyxy[roi*4+1]*scale;
  float x2 = selxyxy[roi*4+2]*scale, y2 = selxyxy[roi*4+3]*scale;
  float bw = (x2-x1)/7.0f, bh = (y2-y1)/7.0f;
  for (int p = threadIdx.x; p < DFEAT; p += blockDim.x){
    int c = p / 49; int rem = p % 49; int iy = rem / 7; int jx = rem % 7;
    const float* fc = feat + (size_t)c * (HFEAT*WFEAT);
    float acc = 0.f;
    #pragma unroll
    for (int sy = 0; sy < 2; ++sy){
      float yv = y1 + ((float)iy + ((float)sy + 0.5f)*0.5f)*bh;
      float yf = floorf(yv); float wy = yv - yf;
      int y0 = (int)yf; y0 = y0 < 0 ? 0 : (y0 > HFEAT-1 ? HFEAT-1 : y0);
      int y1c = (y0+1 > HFEAT-1) ? HFEAT-1 : y0+1;
      #pragma unroll
      for (int sx = 0; sx < 2; ++sx){
        float xv = x1 + ((float)jx + ((float)sx + 0.5f)*0.5f)*bw;
        float xf = floorf(xv); float wx = xv - xf;
        int x0 = (int)xf; x0 = x0 < 0 ? 0 : (x0 > WFEAT-1 ? WFEAT-1 : x0);
        int x1c = (x0+1 > WFEAT-1) ? WFEAT-1 : x0+1;
        float v00 = fc[y0*WFEAT+x0],  v01 = fc[y0*WFEAT+x1c];
        float v10 = fc[y1c*WFEAT+x0], v11 = fc[y1c*WFEAT+x1c];
        acc += v00*(1.f-wy)*(1.f-wx) + v01*(1.f-wy)*wx
             + v10*wy*(1.f-wx)       + v11*wy*wx;
      }
    }
    X[(size_t)roi*DFEAT + p] = f2bf_us(acc * 0.25f);
  }
}

// ---------------- Stage 3: WMMA bf16 GEMM, LDS-staged B ---------------------
// A-frag: 16x32 bf16; lane m=lane&15; K split by lane>>4; one contiguous 32B run (k=16*half+s)
static __device__ __forceinline__ v16bf load_afrag(const unsigned short* arow,
                                                   int k0, int lhalf){
  v16bf a;
  #pragma unroll
  for (int t = 0; t < 8; ++t){
    int k = (t << 1) + (lhalf << 4);
    unsigned pr = *(const unsigned*)(arow + k0 + k);
    a[2*t]   = us_as_bf((unsigned short)(pr & 0xFFFFu));
    a[2*t+1] = us_as_bf((unsigned short)(pr >> 16));
  }
  return a;
}

// Block: 256 threads (8 waves). Covers one 32-wide N panel x all 20 M-tiles.
// wave w: ntile_local = w&1, mgroup = w>>1 (4 groups x 5 M-tiles).
// Per 32-deep k-step: coop-load 32x32 fp32 B tile (1 x f4 per thread, NT hint),
// cvt to bf16, scatter into LDS pre-transposed [nt][n][k] so the B fragment is
// one contiguous 32B v16bf read per lane. Ping-pong buffers, 1 barrier/step.
__global__ __launch_bounds__(256)
void gemm_bf16_kernel(const unsigned short* __restrict__ A,
                      const float* __restrict__ B,
                      const float* __restrict__ bias,
                      unsigned short* __restrict__ Out,
                      int K, int N, int relu){
  __shared__ __align__(64) unsigned short Bs[2][2][16][32];

  int tid   = threadIdx.x;
  int w     = tid >> 5;
  int nt    = w & 1;          // local ntile (0/1)
  int mgrp  = w >> 1;         // 0..3, each owns 5 M-tiles
  int lane  = tid & 31;
  int lhalf = lane >> 4;
  int lmod  = lane & 15;
  int n0    = blockIdx.x * 32;

  // cooperative-loader coordinates
  int krow = tid >> 3;        // 0..31
  int cbase = (tid & 7) * 4;  // 0..28

  v8f acc[5] = {};
  const unsigned short* Ar[5];
  #pragma unroll
  for (int mt = 0; mt < 5; ++mt)
    Ar[mt] = A + (size_t)((mgrp*5 + mt)*16 + lmod) * K;

  int nsteps = K >> 5;
  for (int s = 0; s < nsteps; ++s){
    int k0  = s << 5;
    int buf = s & 1;

    // ---- cooperative B tile load + cvt + transpose-scatter into LDS ----
    f4 bv = __builtin_nontemporal_load(
              (const f4*)(B + (size_t)(k0 + krow)*N + n0 + cbase));
    #pragma unroll
    for (int j = 0; j < 4; ++j){
      int col = cbase + j;
      Bs[buf][col >> 4][col & 15][krow] = f2bf_us(bv[j]);
    }
    __syncthreads();

    // ---- B fragment: one contiguous 32B read per lane ----
    v16bf bfrag = *(const v16bf*)&Bs[buf][nt][lmod][lhalf*16];

    #pragma unroll
    for (int mt = 0; mt < 5; ++mt){
      v16bf af = load_afrag(Ar[mt], k0, lhalf);
      acc[mt] = __builtin_amdgcn_wmma_f32_16x16x32_bf16(
                  false, af, false, bfrag, (short)0, acc[mt], false, false);
    }
  }

  int nidx = n0 + nt*16 + lmod;
  float bvb = bias[nidx];
  #pragma unroll
  for (int mt = 0; mt < 5; ++mt){
    #pragma unroll
    for (int r = 0; r < 8; ++r){
      int m = (mgrp*5 + mt)*16 + r + lhalf*8;
      float v = acc[mt][r] + bvb;
      if (relu) v = fmaxf(v, 0.f);
      Out[(size_t)m * N + nidx] = f2bf_us(v);
    }
  }
}

// ---------------- Stage 4: cls/reg heads + softmax + decode -----------------
__global__ __launch_bounds__(256)
void head_kernel(const unsigned short* __restrict__ H,
                 const float* __restrict__ cls_w, const float* __restrict__ cls_b,
                 const float* __restrict__ reg_w, const float* __restrict__ reg_b,
                 const float* __restrict__ selxywh, float* __restrict__ out){
  __shared__ float xs[HID];
  __shared__ float res[105];
  __shared__ float probs[NCLSD];
  int roi = blockIdx.x; int tid = threadIdx.x;
  for (int i = tid; i < HID; i += 256) xs[i] = bfu_as_f(H[(size_t)roi*HID + i]);
  __syncthreads();
  for (int o = tid; o < 105; o += 256){
    const float* w; int ldw; float acc;
    if (o < NCLSD){ w = cls_w + o;          ldw = NCLSD; acc = cls_b[o]; }
    else          { int j = o - NCLSD; w = reg_w + j; ldw = 84; acc = reg_b[j]; }
    float s = 0.f;
    for (int k = 0; k < HID; ++k) s += xs[k] * w[(size_t)k * ldw];
    acc += s;
    if (o >= NCLSD){ int j = o - NCLSD; acc *= ((j & 3) < 2) ? 0.1f : 0.2f; }
    res[o] = acc;
  }
  __syncthreads();
  if (tid == 0){
    float mx = res[0];
    for (int c = 1; c < NCLSD; ++c) mx = fmaxf(mx, res[c]);
    float sum = 0.f;
    for (int c = 0; c < NCLSD; ++c){ float e = expf(res[c]-mx); probs[c]=e; sum+=e; }
    float inv = 1.f / sum;
    for (int c = 0; c < NCLSD; ++c) probs[c] *= inv;
  }
  __syncthreads();
  if (tid < NCLSD){
    int c = tid;
    out[(size_t)roi*105 + c] = probs[c];
    float px = selxywh[roi*4+0], py = selxywh[roi*4+1];
    float pw = selxywh[roi*4+2], ph = selxywh[roi*4+3];
    float dx = res[NCLSD + c*4+0], dy = res[NCLSD + c*4+1];
    float dw = res[NCLSD + c*4+2], dh = res[NCLSD + c*4+3];
    float nx = px + dx*pw, ny = py + dy*ph;
    float nw = pw * expf(dw), nh = ph * expf(dh);
    float x1 = fminf(fmaxf(nx, 0.f), IMGSZ),      y1 = fminf(fmaxf(ny, 0.f), IMGSZ);
    float x2 = fminf(fmaxf(nx+nw, 0.f), IMGSZ),   y2 = fminf(fmaxf(ny+nh, 0.f), IMGSZ);
    float* od = out + (size_t)roi*105 + NCLSD + c*4;
    od[0]=x1; od[1]=y1; od[2]=x2; od[3]=y2;
  }
}

// ---------------- Launcher ---------------------------------------------------
extern "C" void kernel_launch(void* const* d_in, const int* in_sizes, int n_in,
                              void* d_out, int out_size, void* d_ws, size_t ws_size,
                              hipStream_t stream){
  (void)in_sizes; (void)n_in; (void)out_size; (void)ws_size;
  const float* feat   = (const float*)d_in[0];
  const float* prop   = (const float*)d_in[1];
  const float* scores = (const float*)d_in[2];
  const float* fc1_w  = (const float*)d_in[3];
  const float* fc1_b  = (const float*)d_in[4];
  const float* fc2_w  = (const float*)d_in[5];
  const float* fc2_b  = (const float*)d_in[6];
  const float* cls_w  = (const float*)d_in[7];
  const float* cls_b  = (const float*)d_in[8];
  const float* reg_w  = (const float*)d_in[9];
  const float* reg_b  = (const float*)d_in[10];
  float* out = (float*)d_out;

  char* ws = (char*)d_ws;
  size_t off = 0;
  auto carve = [&](size_t bytes)->char*{
    char* p = ws + off;
    off += (bytes + 255) & ~(size_t)255;
    return p;
  };
  float*          sc      = (float*)carve((size_t)NPROP*4);
  float*          bxyxy   = (float*)carve((size_t)NPROP*16);
  int*            keep    = (int*)  carve((size_t)POSTN*4);
  float*          selxywh = (float*)carve((size_t)POSTN*16);
  float*          selxyxy = (float*)carve((size_t)POSTN*16);
  unsigned short* X       = (unsigned short*)carve((size_t)MPAD*DFEAT*2);
  unsigned short* H1      = (unsigned short*)carve((size_t)MPAD*HID*2);
  unsigned short* H2      = (unsigned short*)carve((size_t)MPAD*HID*2);

  prep_kernel<<<(NPROP+255)/256, 256, 0, stream>>>(prop, scores, sc, bxyxy);
  nms_kernel<<<1, 1024, 0, stream>>>(bxyxy, prop, sc, keep, selxywh, selxyxy);
  roi_kernel<<<POSTN, 256, 0, stream>>>(feat, selxyxy, X);
  // fc1: M=320 (20 M-tiles), N=4096 -> 128 blocks of 32 cols, K=25088
  gemm_bf16_kernel<<<HID/32, 256, 0, stream>>>(X,  fc1_w, fc1_b, H1, DFEAT, HID, 1);
  // fc2: K=4096
  gemm_bf16_kernel<<<HID/32, 256, 0, stream>>>(H1, fc2_w, fc2_b, H2, HID,   HID, 1);
  head_kernel<<<POSTN, 256, 0, stream>>>(H2, cls_w, cls_b, reg_w, reg_b, selxywh, out);
}
